// Block_43774306681140
// MI455X (gfx1250) — compile-verified, matrix operand order checked by reading, not run on hardware
//
#include <hip/hip_runtime.h>
#include <math.h>

// ---------------------------------------------------------------------------
// CDNA5 (gfx1250) wave32 WMMA types
// ---------------------------------------------------------------------------
typedef __attribute__((ext_vector_type(16))) __bf16 v16bf;
typedef __attribute__((ext_vector_type(8)))  __bf16 v8bf;
typedef __attribute__((ext_vector_type(4)))  __bf16 v4bf;
typedef __attribute__((ext_vector_type(8)))  float  v8f;
typedef __attribute__((ext_vector_type(4)))  float  v4f;

#define WMMA_BF16(a, b, c) \
  __builtin_amdgcn_wmma_f32_16x16x32_bf16(false, (a), false, (b), (short)0, (c), false, false)

// ---------------------------------------------------------------------------
// Async global -> LDS copy (CDNA5 GLOBAL_LOAD_ASYNC_TO_LDS_B128, ASYNCcnt).
// Low 32 bits of a flat pointer into __shared__ are the wave-relative LDS
// byte offset (ISA 10.2: LDS aperture).
// ---------------------------------------------------------------------------
#define USE_ASYNC_LDS 1

__device__ __forceinline__ void async_cp16(void* lds, const void* g) {
#if USE_ASYNC_LDS
  asm volatile("global_load_async_to_lds_b128 %0, %1, off"
               :: "v"((unsigned)(unsigned long long)lds),
                  "v"((unsigned long long)g)
               : "memory");
#else
  *(v8bf*)lds = *(const v8bf*)g;
#endif
}
__device__ __forceinline__ void wait_async() {
#if USE_ASYNC_LDS
  asm volatile("s_wait_asynccnt 0x0" ::: "memory");
#endif
}

// Load one 16x32 bf16 fragment (A-style or B-style; "row" = M for A, N for B)
// from an LDS tile stored row-major.  ISA 7.12.2 layout: lanes 0-15 hold
// row=lane, K=0..7 & 16..23; lanes 16-31 hold row=lane-16, K=8..15 & 24..31.
__device__ __forceinline__ v16bf frag_ld(const __bf16* base, int stride) {
  const int lane = threadIdx.x & 31;
  const int r    = lane & 15;
  const int ko   = (lane >> 4) << 3;
  v8bf lo = *(const v8bf*)(base + r * stride + ko);
  v8bf hi = *(const v8bf*)(base + r * stride + ko + 16);
  v16bf o;
#pragma unroll
  for (int i = 0; i < 8; ++i) { o[i] = lo[i]; o[i + 8] = hi[i]; }
  return o;
}

// ---------------------------------------------------------------------------
// WMMA GEMM:  C[M,N] = A[M,K](bf16, row-major) @ B[N,K](f32 weights)^T
// Block tile 128x128, K-step 32, 256 threads = 8 waves, wave tile 32x64.
// Double-buffered LDS: A tile staged by async-DMA, B tile staged through
// VGPRs with on-the-fly f32->bf16 conversion, overlapped with the WMMAs.
// epi: 0 = store f32, 1 = residual + gate[col]*acc (f32), 2 = GELU -> bf16
// ---------------------------------------------------------------------------
#define GBM 128
#define GBN 128
#define GBK 32
#define GLD 40   // padded LDS row stride (elements); 80B rows keep 16B align

__global__ __launch_bounds__(256) void wmma_gemm(
    const __bf16* __restrict__ A, const float* __restrict__ Bw,
    float* __restrict__ Cf, __bf16* __restrict__ Cb,
    const float* __restrict__ Res, const float* __restrict__ Gate,
    int M, int N, int K, int epi)
{
  __shared__ __align__(16) __bf16 As[2][GBM * GLD];
  __shared__ __align__(16) __bf16 Bs[2][GBN * GLD];

  const int tid  = threadIdx.x;
  const int lane = tid & 31;
  const int wid  = tid >> 5;
  const int wm   = (wid >> 1) * 32;   // 4 waves along M
  const int wn   = (wid & 1) * 64;    // 2 waves along N
  const int bm   = blockIdx.y * GBM;
  const int bn   = blockIdx.x * GBN;

  // Per-thread staging coordinates.
  const int ar0 = (tid * 2) >> 2,       ac0 = ((tid * 2) & 3) << 3;       // A chunk 0
  const int ar1 = (tid * 2 + 1) >> 2,   ac1 = ((tid * 2 + 1) & 3) << 3;   // A chunk 1

  v8f acc[2][4] = {};
  v4f breg[4];

  auto stage_a = [&](int buf, int k0) {
    async_cp16(&As[buf][ar0 * GLD + ac0], A + (size_t)(bm + ar0) * K + k0 + ac0);
    async_cp16(&As[buf][ar1 * GLD + ac1], A + (size_t)(bm + ar1) * K + k0 + ac1);
  };
  auto load_b = [&](int k0) {
#pragma unroll
    for (int c = 0; c < 4; ++c) {
      int chunk = tid * 4 + c;
      int row = chunk >> 3, cc = (chunk & 7) << 2;
      breg[c] = *(const v4f*)(Bw + (size_t)(bn + row) * K + k0 + cc);
    }
  };
  auto store_b = [&](int buf) {
#pragma unroll
    for (int c = 0; c < 4; ++c) {
      int chunk = tid * 4 + c;
      int row = chunk >> 3, cc = (chunk & 7) << 2;
      v4bf h;
      h[0] = (__bf16)breg[c][0]; h[1] = (__bf16)breg[c][1];
      h[2] = (__bf16)breg[c][2]; h[3] = (__bf16)breg[c][3];
      *(v4bf*)&Bs[buf][row * GLD + cc] = h;
    }
  };

  // Prologue: fill buffer 0.
  stage_a(0, 0);
  load_b(0);
  store_b(0);
  wait_async();
  __syncthreads();

  int buf = 0;
  for (int k0 = 0; k0 < K; k0 += GBK) {
    const bool has_next = (k0 + GBK) < K;
    if (has_next) {
      stage_a(buf ^ 1, k0 + GBK);   // DMA next A tile while we compute
      load_b(k0 + GBK);             // next B tile in flight in VGPRs
    }

    v16bf af[2], bfr[4];
#pragma unroll
    for (int i = 0; i < 2; ++i) af[i]  = frag_ld(&As[buf][(wm + i * 16) * GLD], GLD);
#pragma unroll
    for (int j = 0; j < 4; ++j) bfr[j] = frag_ld(&Bs[buf][(wn + j * 16) * GLD], GLD);
#pragma unroll
    for (int i = 0; i < 2; ++i)
#pragma unroll
      for (int j = 0; j < 4; ++j)
        acc[i][j] = WMMA_BF16(af[i], bfr[j], acc[i][j]);

    if (has_next) store_b(buf ^ 1); // convert + commit after WMMAs issued
    wait_async();                   // publish our async A writes
    __syncthreads();
    buf ^= 1;
  }

  // C layout (ISA 7.12.2): VGPR r -> M = r (lanes 0-15) / r+8 (lanes 16-31),
  // N = lane & 15.
  const int half8 = (lane >> 4) * 8;
  const int cn    = lane & 15;
#pragma unroll
  for (int i = 0; i < 2; ++i)
#pragma unroll
    for (int j = 0; j < 4; ++j)
#pragma unroll
      for (int r = 0; r < 8; ++r) {
        int grow = bm + wm + i * 16 + r + half8;
        int gcol = bn + wn + j * 16 + cn;
        size_t idx = (size_t)grow * N + gcol;
        float v = acc[i][j][r];
        if (epi == 0)      Cf[idx] = v;
        else if (epi == 1) Cf[idx] = Res[idx] + Gate[gcol] * v;
        else               Cb[idx] = (__bf16)(0.5f * v * (1.0f + erff(v * 0.70710678118f)));
      }
}

// ---------------------------------------------------------------------------
// Flash attention (one head, 64 queries per block, 4 waves, HD = 64).
// Q,K,V,O are bf16 (seq, 16*64).  Online softmax; scores & P*V on WMMA.
// Q and K tiles staged by async-DMA; V is transposed into [d][key] manually.
// ---------------------------------------------------------------------------
__global__ __launch_bounds__(128) void fa_kernel(
    const __bf16* __restrict__ Q, const __bf16* __restrict__ Kp,
    const __bf16* __restrict__ Vp, __bf16* __restrict__ O,
    int Skv, float scale)
{
  __shared__ __align__(16) __bf16 Qs[64 * 72];
  __shared__ __align__(16) __bf16 Ks[64 * 72];
  __shared__ __align__(16) __bf16 Vt[64 * 72];   // transposed: [d][key]
  __shared__ __align__(16) __bf16 Ps[64 * 72];
  __shared__ float Sf[64 * 68];
  __shared__ float rowm[64], rowl[64], rowa[64];

  const int tid   = threadIdx.x;
  const int lane  = tid & 31;
  const int w     = tid >> 5;
  const int half8 = (lane >> 4) * 8;
  const int cn    = lane & 15;
  const int head  = blockIdx.y;
  const int qb    = blockIdx.x * 64;
  const size_t hoff = (size_t)head * 64;

  // Load Q tile (64x64) via async DMA, 4 16B chunks per thread.
#pragma unroll
  for (int c = 0; c < 4; ++c) {
    int chunk = tid * 4 + c;
    int row = chunk >> 3, cc = (chunk & 7) << 3;
    async_cp16(&Qs[row * 72 + cc], Q + (size_t)(qb + row) * 1024 + hoff + cc);
  }
  if (tid < 64) { rowm[tid] = -1e30f; rowl[tid] = 0.0f; }
  v8f accO[4] = {};

  for (int kb = 0; kb < Skv; kb += 64) {
    __syncthreads();
    // K tile: async copy.  V tile: load + transpose to [d][key].
#pragma unroll
    for (int c = 0; c < 4; ++c) {
      int chunk = tid * 4 + c;
      int row = chunk >> 3, cc = (chunk & 7) << 3;
      async_cp16(&Ks[row * 72 + cc], Kp + (size_t)(kb + row) * 1024 + hoff + cc);
      v8bf vv = *(const v8bf*)(Vp + (size_t)(kb + row) * 1024 + hoff + cc);
#pragma unroll
      for (int j = 0; j < 8; ++j) Vt[(cc + j) * 72 + row] = vv[j];
    }
    wait_async();
    __syncthreads();

    // Scores: wave w -> queries [w*16, w*16+16) vs 64 keys; HD=64 -> 2 K-steps.
#pragma unroll
    for (int n = 0; n < 4; ++n) {
      v8f s = {};
#pragma unroll
      for (int ks = 0; ks < 64; ks += 32) {
        v16bf a = frag_ld(&Qs[(w * 16) * 72 + ks], 72);
        v16bf b = frag_ld(&Ks[(n * 16) * 72 + ks], 72);
        s = WMMA_BF16(a, b, s);
      }
#pragma unroll
      for (int r = 0; r < 8; ++r)
        Sf[(w * 16 + r + half8) * 68 + n * 16 + cn] = s[r] * scale;
    }
    __syncthreads();

    // Online softmax: one thread per query row.
    if (tid < 64) {
      float mo = rowm[tid], mx = mo;
      for (int j = 0; j < 64; ++j) mx = fmaxf(mx, Sf[tid * 68 + j]);
      float al = __expf(mo - mx), ls = 0.0f;
      for (int j = 0; j < 64; ++j) {
        float p = __expf(Sf[tid * 68 + j] - mx);
        Ps[tid * 72 + j] = (__bf16)p;
        ls += p;
      }
      rowm[tid] = mx;
      rowl[tid] = rowl[tid] * al + ls;
      rowa[tid] = al;
    }
    __syncthreads();

    // Rescale accumulators and add P @ V.
#pragma unroll
    for (int n = 0; n < 4; ++n) {
#pragma unroll
      for (int r = 0; r < 8; ++r) accO[n][r] *= rowa[w * 16 + r + half8];
#pragma unroll
      for (int ks = 0; ks < 64; ks += 32) {
        v16bf a = frag_ld(&Ps[(w * 16) * 72 + ks], 72);
        v16bf b = frag_ld(&Vt[(n * 16) * 72 + ks], 72);
        accO[n] = WMMA_BF16(a, b, accO[n]);
      }
    }
  }
  __syncthreads();

#pragma unroll
  for (int n = 0; n < 4; ++n)
#pragma unroll
    for (int r = 0; r < 8; ++r) {
      int row = w * 16 + r + half8;
      float inv = 1.0f / rowl[row];
      O[(size_t)(qb + row) * 1024 + hoff + n * 16 + cn] =
          (__bf16)(accO[n][r] * inv);
    }
}

// ---------------------------------------------------------------------------
// Per-(seq, head) RMSNorm + RoPE -> bf16.  One wave per 64-elem group.
// Wn == nullptr: skip norm.  Rope == nullptr: skip rotation (also used as a
// plain f32 -> bf16 converter).
// ---------------------------------------------------------------------------
__global__ __launch_bounds__(256) void norm_rope_kernel(
    const float* __restrict__ X, const float* __restrict__ Wn,
    const float* __restrict__ Rope, __bf16* __restrict__ Y, int rows)
{
  int g    = blockIdx.x * 8 + (threadIdx.x >> 5);
  int lane = threadIdx.x & 31;
  int row  = g >> 4;
  int head = g & 15;
  const float* xp = X + (size_t)row * 1024 + head * 64;
  float x1 = xp[lane], x2 = xp[lane + 32];
  float y1 = x1, y2 = x2;
  if (Wn) {
    float ss = x1 * x1 + x2 * x2;
#pragma unroll
    for (int o = 16; o; o >>= 1) ss += __shfl_xor(ss, o, 32);
    float r = rsqrtf(ss * (1.0f / 64.0f) + 1e-6f);
    y1 = x1 * r * Wn[lane];
    y2 = x2 * r * Wn[lane + 32];
  }
  if (Rope) {
    float c = Rope[(size_t)row * 64 + lane];
    float s = Rope[(size_t)row * 64 + 32 + lane];
    float o1 = y1 * c - y2 * s;
    float o2 = y2 * c + y1 * s;
    y1 = o1; y2 = o2;
  }
  __bf16* yp = Y + (size_t)row * 1024 + head * 64;
  yp[lane]      = (__bf16)y1;
  yp[lane + 32] = (__bf16)y2;
}

// ---------------------------------------------------------------------------
// LayerNorm(x) * (1 + scale) + shift  -> bf16.  One 256-thread block per row.
// ---------------------------------------------------------------------------
__global__ __launch_bounds__(256) void ln_mod_kernel(
    const float* __restrict__ X, const float* __restrict__ Sh,
    const float* __restrict__ Sc, __bf16* __restrict__ H)
{
  __shared__ float s1[8], s2[8];
  int row = blockIdx.x, tid = threadIdx.x, lane = tid & 31, wid = tid >> 5;
  v4f v = *(const v4f*)(X + (size_t)row * 1024 + tid * 4);
  float sum = v[0] + v[1] + v[2] + v[3];
  float ssq = v[0] * v[0] + v[1] * v[1] + v[2] * v[2] + v[3] * v[3];
#pragma unroll
  for (int o = 16; o; o >>= 1) {
    sum += __shfl_xor(sum, o, 32);
    ssq += __shfl_xor(ssq, o, 32);
  }
  if (lane == 0) { s1[wid] = sum; s2[wid] = ssq; }
  __syncthreads();
  float ts = 0.f, tq = 0.f;
#pragma unroll
  for (int i = 0; i < 8; ++i) { ts += s1[i]; tq += s2[i]; }
  float mean = ts * (1.0f / 1024.0f);
  float var  = tq * (1.0f / 1024.0f) - mean * mean;
  float rstd = rsqrtf(var + 1e-6f);
#pragma unroll
  for (int j = 0; j < 4; ++j) {
    int c = tid * 4 + j;
    H[(size_t)row * 1024 + c] =
        (__bf16)((v[j] - mean) * rstd * (1.0f + Sc[c]) + Sh[c]);
  }
}

// ---------------------------------------------------------------------------
// silu(emb) and the three tiny modulation GEMVs.
// ---------------------------------------------------------------------------
__global__ void silu_kernel(const float* __restrict__ E, float* __restrict__ S) {
  int i = blockIdx.x * 256 + threadIdx.x;
  float e = E[i];
  S[i] = e / (1.0f + __expf(-e));
}

__global__ void mod_gemv_kernel(
    const float* __restrict__ Se,
    const float* __restrict__ W0, const float* __restrict__ W1,
    const float* __restrict__ W2, float* __restrict__ Out)
{
  int o = blockIdx.x * 256 + threadIdx.x;        // 0..9215
  int widx = o / 3072, j = o % 3072;
  const float* w = (widx == 0 ? W0 : (widx == 1 ? W1 : W2)) + (size_t)j * 1024;
  float acc = 0.f;
  for (int k = 0; k < 1024; k += 4) {
    v4f a = *(const v4f*)(Se + k);
    v4f b = *(const v4f*)(w + k);
    acc += a[0] * b[0] + a[1] * b[1] + a[2] * b[2] + a[3] * b[3];
  }
  Out[o] = acc;
}

// ---------------------------------------------------------------------------
// Host orchestration.
// ---------------------------------------------------------------------------
extern "C" void kernel_launch(void* const* d_in, const int* in_sizes, int n_in,
                              void* d_out, int out_size, void* d_ws, size_t ws_size,
                              hipStream_t stream) {
  (void)in_sizes; (void)n_in; (void)out_size; (void)ws_size;
  const float* x     = (const float*)d_in[0];
  const float* emb   = (const float*)d_in[1];
  const float* ctx   = (const float*)d_in[2];
  const float* sa_wq = (const float*)d_in[3];
  const float* sa_wk = (const float*)d_in[4];
  const float* sa_wv = (const float*)d_in[5];
  const float* sa_wo = (const float*)d_in[6];
  const float* sa_qn = (const float*)d_in[7];
  const float* sa_kn = (const float*)d_in[8];
  const float* ca_wq = (const float*)d_in[9];
  const float* ca_wk = (const float*)d_in[10];
  const float* ca_wv = (const float*)d_in[11];
  const float* ca_wo = (const float*)d_in[12];
  const float* ca_qn = (const float*)d_in[13];
  const float* ca_kn = (const float*)d_in[14];
  const float* w1    = (const float*)d_in[15];
  const float* w2    = (const float*)d_in[16];
  const float* mw_sa = (const float*)d_in[17];
  const float* mw_ca = (const float*)d_in[18];
  const float* mw_ml = (const float*)d_in[19];
  const float* rope  = (const float*)d_in[20];
  float* out = (float*)d_out;

  const int S = 4096, D = 1024, Mkv = 512, DF = 4096;

  char* ws = (char*)d_ws;
  size_t off = 0;
  auto alloc = [&](size_t bytes) {
    void* p = ws + off;
    off += (bytes + 255) & ~(size_t)255;
    return p;
  };
  float*  silu_e = (float*) alloc(1024 * 4);
  float*  modb   = (float*) alloc(9216 * 4);
  __bf16* h_bf   = (__bf16*)alloc((size_t)S * D * 2);
  float*  qkvf   = (float*) alloc((size_t)S * D * 4);
  __bf16* q_bf   = (__bf16*)alloc((size_t)S * D * 2);
  __bf16* k_bf   = (__bf16*)alloc((size_t)S * D * 2);
  __bf16* v_bf   = (__bf16*)alloc((size_t)S * D * 2);
  __bf16* ao_bf  = (__bf16*)alloc((size_t)S * D * 2);
  __bf16* ctx_bf = (__bf16*)alloc((size_t)Mkv * D * 2);
  __bf16* hid_bf = (__bf16*)alloc((size_t)S * DF * 2);

  const dim3 blk(256);
  const dim3 ggD(D / 128, S / 128);        // 4096 x 1024 output
  const dim3 ggK(D / 128, Mkv / 128);      // 512  x 1024 output
  const dim3 ggF(DF / 128, S / 128);       // 4096 x 4096 output
  const dim3 gfa(S / 64, 16);

  // modulation
  silu_kernel<<<4, blk, 0, stream>>>(emb, silu_e);
  mod_gemv_kernel<<<36, blk, 0, stream>>>(silu_e, mw_sa, mw_ca, mw_ml, modb);

  // ---- self attention ----
  ln_mod_kernel<<<S, blk, 0, stream>>>(x, modb + 0, modb + 1024, h_bf);
  wmma_gemm<<<ggD, blk, 0, stream>>>(h_bf, sa_wq, qkvf, nullptr, nullptr, nullptr, S, D, D, 0);
  norm_rope_kernel<<<S * 16 / 8, blk, 0, stream>>>(qkvf, sa_qn, rope, q_bf, S);
  wmma_gemm<<<ggD, blk, 0, stream>>>(h_bf, sa_wk, qkvf, nullptr, nullptr, nullptr, S, D, D, 0);
  norm_rope_kernel<<<S * 16 / 8, blk, 0, stream>>>(qkvf, sa_kn, rope, k_bf, S);
  wmma_gemm<<<ggD, blk, 0, stream>>>(h_bf, sa_wv, qkvf, nullptr, nullptr, nullptr, S, D, D, 0);
  norm_rope_kernel<<<S * 16 / 8, blk, 0, stream>>>(qkvf, nullptr, nullptr, v_bf, S);
  fa_kernel<<<gfa, dim3(128), 0, stream>>>(q_bf, k_bf, v_bf, ao_bf, S, 0.125f);
  wmma_gemm<<<ggD, blk, 0, stream>>>(ao_bf, sa_wo, out, nullptr, x, modb + 2048, S, D, D, 1);

  // ---- cross attention ----
  ln_mod_kernel<<<S, blk, 0, stream>>>(out, modb + 3072, modb + 3072 + 1024, h_bf);
  wmma_gemm<<<ggD, blk, 0, stream>>>(h_bf, ca_wq, qkvf, nullptr, nullptr, nullptr, S, D, D, 0);
  norm_rope_kernel<<<S * 16 / 8, blk, 0, stream>>>(qkvf, ca_qn, nullptr, q_bf, S);
  norm_rope_kernel<<<Mkv * 16 / 8, blk, 0, stream>>>(ctx, nullptr, nullptr, ctx_bf, Mkv);
  wmma_gemm<<<ggK, blk, 0, stream>>>(ctx_bf, ca_wk, qkvf, nullptr, nullptr, nullptr, Mkv, D, D, 0);
  norm_rope_kernel<<<Mkv * 16 / 8, blk, 0, stream>>>(qkvf, ca_kn, nullptr, k_bf, Mkv);
  wmma_gemm<<<ggK, blk, 0, stream>>>(ctx_bf, ca_wv, qkvf, nullptr, nullptr, nullptr, Mkv, D, D, 0);
  norm_rope_kernel<<<Mkv * 16 / 8, blk, 0, stream>>>(qkvf, nullptr, nullptr, v_bf, Mkv);
  fa_kernel<<<gfa, dim3(128), 0, stream>>>(q_bf, k_bf, v_bf, ao_bf, Mkv, 0.125f);
  wmma_gemm<<<ggD, blk, 0, stream>>>(ao_bf, ca_wo, out, nullptr, out, modb + 3072 + 2048, S, D, D, 1);

  // ---- MLP ----
  ln_mod_kernel<<<S, blk, 0, stream>>>(out, modb + 6144, modb + 6144 + 1024, h_bf);
  wmma_gemm<<<ggF, blk, 0, stream>>>(h_bf, w1, nullptr, hid_bf, nullptr, nullptr, S, DF, D, 2);
  wmma_gemm<<<ggD, blk, 0, stream>>>(hid_bf, w2, out, nullptr, out, modb + 6144 + 2048, S, D, DF, 1);
}
